// FastShiftNet_32512902431268
// MI455X (gfx1250) — matrix operand order repaired and checked by reading, not compiled, not run on hardware
//
#include <hip/hip_runtime.h>
#include <hip/hip_bf16.h>
#include <math.h>

// ---------------------------------------------------------------------------
// FastShiftNet fused kernel for MI455X (gfx1250), wave32 + WMMA bf16 + TDM.
//   out[b,i] = sum_{s<=i} softmax(MLP(shift_bits[b]))[s] * a_bits[b, i-s]
// ---------------------------------------------------------------------------

#define BATCH 65536
#define BITS  64
#define HID   512
#define ROWS  64          // batch rows per workgroup
#define NTH   512         // 16 waves
#define SA_STRIDE 520     // bf16 elems, padded (1040B rows -> conflict-free frags)
#define SB_STRIDE 72      // bf16 elems, padded (144B rows)
#define SA1_STRIDE 72     // GEMM1 A tile stride (K=64)
#define SL_STRIDE 68      // fp32 logits stride
#define SP_STRIDE 72      // fp32 probs / a_bits stride

typedef __attribute__((ext_vector_type(16))) __bf16 v16bf;
typedef __attribute__((ext_vector_type(8)))  float  v8f;

union FragU { v16bf v; uint4 q[2]; };

#if defined(__AMDGCN__) && __has_builtin(__builtin_amdgcn_tensor_load_to_lds) && \
    __has_builtin(__builtin_amdgcn_s_wait_tensorcnt)
#define HAVE_TDM 1
#else
#define HAVE_TDM 0
#endif

#if HAVE_TDM
typedef unsigned int u32x4 __attribute__((ext_vector_type(4)));
typedef int          i32x4 __attribute__((ext_vector_type(4)));
typedef int          i32x8 __attribute__((ext_vector_type(8)));
typedef __attribute__((address_space(3))) const char* lds_cptr_t;

__device__ __forceinline__ unsigned lds_off_u32(const void* p) {
    return (unsigned)(unsigned long long)(lds_cptr_t)p;
}

// Issue one 2D TDM tile load (bf16 elements) with LDS row padding.
//   pad: after every 8*2^pad_interval bytes stored, skip (pad_amount+1) DWORDs.
__device__ __forceinline__ void tdm_load_2d_bf16(const void* gsrc, unsigned lds_off,
                                                 int tensor_d0, int tensor_d1,
                                                 int tile_d0, int tile_d1, int stride0,
                                                 int pad_interval, int pad_amount) {
    unsigned long long ga = (unsigned long long)gsrc;
    u32x4 g0 = {
        1u,                                              // count=1 (valid descriptor)
        lds_off,                                         // lds_addr
        (unsigned)(ga & 0xFFFFFFFFu),                    // global_addr[31:0]
        (unsigned)((ga >> 32) & 0x1FFFFFFu) | (2u << 30) // global_addr[56:32] | type=2
    };
    i32x8 g1 = {
        (1 << 16) | (1 << 20) | (pad_interval << 22) | (pad_amount << 25),
        (tensor_d0 & 0xFFFF) << 16,                              // tensor_dim0[15:0]
        ((tensor_d0 >> 16) & 0xFFFF) | ((tensor_d1 & 0xFFFF) << 16),
        ((tensor_d1 >> 16) & 0xFFFF) | ((tile_d0 & 0xFFFF) << 16),
        (tile_d1 & 0xFFFF),                                      // tile_dim1, tile_dim2=0
        stride0,                                                 // tensor_dim0_stride
        0, 0
    };
    i32x4 gz4 = {0, 0, 0, 0};
    i32x8 gz8 = {0, 0, 0, 0, 0, 0, 0, 0};
    // amdgpu-toolchain (clang-23) 6-arg form: (g0, g1, g2, g3, extra, cpol)
    __builtin_amdgcn_tensor_load_to_lds(g0, g1, gz4, gz4, gz8, 0);
}
#define WAIT_TENSOR(n) __builtin_amdgcn_s_wait_tensorcnt(n)
#else
#define WAIT_TENSOR(n) ((void)0)
#endif

__device__ __forceinline__ v8f wmma_bf16(v16bf a, v16bf b, v8f c) {
    return __builtin_amdgcn_wmma_f32_16x16x32_bf16(
        false, a, false, b, (short)0, c, false, false);
}

// A fragment 16x32 bf16 (ISA 7.12.2): lane L<16 row M=L holds K {0..7,16..23},
// lane L>=16 row M=L-16 holds K {8..15,24..31} => two contiguous 16B runs.
__device__ __forceinline__ v16bf load_frag_a(const __bf16* base, int stride,
                                             int row0, int kbase, int lane) {
    const int r  = row0 + (lane & 15);
    const int kh = (lane >> 4) & 1;
    const __bf16* p = base + r * stride + kbase + kh * 8;
    FragU f;
    f.q[0] = *(const uint4*)(p);
    f.q[1] = *(const uint4*)(p + 16);
    return f.v;
}

// B fragment 32x16 bf16 from N-major (transposed) tile: lane = column,
// lane<16 K {0..7,16..23}, lane>=16 K {8..15,24..31} => two contiguous 16B runs.
__device__ __forceinline__ v16bf load_frag_b(const __bf16* baseT, int stride,
                                             int col0, int kbase, int lane) {
    const int c  = col0 + (lane & 15);
    const int kh = (lane >> 4) & 1;
    const __bf16* p = baseT + c * stride + kbase + kh * 8;
    FragU f;
    f.q[0] = *(const uint4*)(p);
    f.q[1] = *(const uint4*)(p + 16);
    return f.v;
}

// C/D 16x16 f32 layout: VGPR i, lanes 0..15 -> (M=i, N=lane); lanes 16..31 -> (M=8+i).
__device__ __forceinline__ void store_acc_lds(float* base, int stride, int row0,
                                              int col0, v8f acc, int lane) {
    const int n  = col0 + (lane & 15);
    const int mo = (lane >> 4) * 8;
#pragma unroll
    for (int i = 0; i < 8; ++i) base[(row0 + mo + i) * stride + n] = acc[i];
}

__device__ __forceinline__ float gelu_exact(float x) {
    return 0.5f * x * (1.0f + erff(x * 0.70710678118654752f));
}

// ---- weight staging: TDM DMA (wave 0) or cooperative-copy fallback ----------
__device__ __forceinline__ void issue_w1(__bf16* dst, const __bf16* w1t,
                                         int tid, int wave) {
#if HAVE_TDM
    if (wave == 0)
        tdm_load_2d_bf16(w1t, lds_off_u32(dst), BITS, HID, BITS, HID, BITS, 4, 3);
#else
    for (int v = tid; v < HID * BITS / 8; v += NTH) {
        int n = v >> 3, k8 = (v & 7) * 8;
        *(uint4*)(dst + n * SB_STRIDE + k8) = *(const uint4*)(w1t + n * BITS + k8);
    }
#endif
}

__device__ __forceinline__ void issue_w2_chunk(__bf16* dst, const __bf16* w2t,
                                               int kc, int tid, int wave) {
#if HAVE_TDM
    if (wave == 0)
        tdm_load_2d_bf16(w2t + kc * 64, lds_off_u32(dst), HID, HID, 64, HID, HID, 4, 3);
#else
    for (int v = tid; v < HID * 64 / 8; v += NTH) {
        int n = v >> 3, k8 = (v & 7) * 8;
        *(uint4*)(dst + n * SB_STRIDE + k8) =
            *(const uint4*)(w2t + n * HID + kc * 64 + k8);
        __builtin_prefetch(w2t + n * HID + ((kc + 1) & 7) * 64 + k8, 0, 1);
    }
#endif
}

__device__ __forceinline__ void issue_w3(__bf16* dst, const __bf16* w3t,
                                         int tid, int wave) {
#if HAVE_TDM
    if (wave == 0)
        tdm_load_2d_bf16(w3t, lds_off_u32(dst), HID, BITS, HID, BITS, HID, 7, 3);
#else
    for (int v = tid; v < BITS * HID / 8; v += NTH) {
        int n = v >> 6, k8 = (v & 63) * 8;
        *(uint4*)(dst + n * SA_STRIDE + k8) = *(const uint4*)(w3t + n * HID + k8);
    }
#endif
}

// In-register LayerNorm + GELU over a wave's 16x128 accumulator patch,
// cross-wave combine through a tiny LDS array, result written as bf16 A-tile.
__device__ __forceinline__ void ln_gelu_to_sA(v8f acc[8],
                                              const float* __restrict__ bias,
                                              const float* __restrict__ gain,
                                              const float* __restrict__ beta,
                                              __bf16* sA, int saStride,
                                              int wm, int wn, int lane,
                                              float* sRedS, float* sRedQ) {
    const int l15 = lane & 15;
    const int mo  = (lane >> 4) * 8;
    const int colBase = wn * 128;
#pragma unroll
    for (int nt = 0; nt < 8; ++nt) {
        float bc = bias[colBase + nt * 16 + l15];
#pragma unroll
        for (int i = 0; i < 8; ++i) acc[nt][i] += bc;
    }
    float s[8], q[8];
#pragma unroll
    for (int i = 0; i < 8; ++i) {
        float ss = 0.f, qq = 0.f;
#pragma unroll
        for (int nt = 0; nt < 8; ++nt) { float x = acc[nt][i]; ss += x; qq += x * x; }
#pragma unroll
        for (int m = 1; m < 16; m <<= 1) {
            ss += __shfl_xor(ss, m, 16);
            qq += __shfl_xor(qq, m, 16);
        }
        s[i] = ss; q[i] = qq;
    }
    if (l15 == 0) {
#pragma unroll
        for (int i = 0; i < 8; ++i) {
            int row = wm * 16 + mo + i;
            sRedS[row * 4 + wn] = s[i];
            sRedQ[row * 4 + wn] = q[i];
        }
    }
    __syncthreads();
    float mean[8], inv[8];
#pragma unroll
    for (int i = 0; i < 8; ++i) {
        int row = wm * 16 + mo + i;
        float S = sRedS[row * 4 + 0] + sRedS[row * 4 + 1] + sRedS[row * 4 + 2] + sRedS[row * 4 + 3];
        float Q = sRedQ[row * 4 + 0] + sRedQ[row * 4 + 1] + sRedQ[row * 4 + 2] + sRedQ[row * 4 + 3];
        float m = S * (1.0f / 512.0f);
        float v = Q * (1.0f / 512.0f) - m * m;
        mean[i] = m;
        inv[i]  = rsqrtf(v + 1e-5f);
    }
#pragma unroll
    for (int nt = 0; nt < 8; ++nt) {
        int col = colBase + nt * 16 + l15;
        float gc = gain[col], bc = beta[col];
#pragma unroll
        for (int i = 0; i < 8; ++i) {
            int row = wm * 16 + mo + i;
            float y = (acc[nt][i] - mean[i]) * inv[i] * gc + bc;
            sA[row * saStride + col] = (__bf16)gelu_exact(y);
        }
    }
}

// ---------------------------------------------------------------------------
// One-shot fp32 -> bf16 weight conversion, transposed to [N][K] in workspace.
//   ws layout (bf16 elems): W1T[512][64] | W2T[512][512] | W3T[64][512]
// ---------------------------------------------------------------------------
__global__ void convert_weights_kernel(const float* __restrict__ W1,
                                       const float* __restrict__ W2,
                                       const float* __restrict__ W3,
                                       __bf16* __restrict__ ws) {
    const int i0 = blockIdx.x * blockDim.x + threadIdx.x;
    const int stride = gridDim.x * blockDim.x;
    __bf16* w1t = ws;
    __bf16* w2t = ws + BITS * HID;
    __bf16* w3t = ws + BITS * HID + HID * HID;
    for (int idx = i0; idx < BITS * HID; idx += stride) {
        int n = idx >> 6, k = idx & 63;
        w1t[idx] = (__bf16)W1[k * HID + n];
    }
    for (int idx = i0; idx < HID * HID; idx += stride) {
        int n = idx >> 9, k = idx & 511;
        w2t[idx] = (__bf16)W2[k * HID + n];
    }
    for (int idx = i0; idx < BITS * HID; idx += stride) {
        int n = idx >> 9, k = idx & 511;
        w3t[idx] = (__bf16)W3[k * BITS + n];
    }
}

// ---------------------------------------------------------------------------
// Fully fused MLP + softmax + causal-shift-conv kernel.
// ---------------------------------------------------------------------------
__global__ __launch_bounds__(NTH, 1)
void fastshiftnet_fused(const float* __restrict__ a_bits,
                        const float* __restrict__ shift_bits,
                        const float* __restrict__ b1, const float* __restrict__ g1,
                        const float* __restrict__ be1,
                        const float* __restrict__ b2, const float* __restrict__ g2,
                        const float* __restrict__ be2,
                        const float* __restrict__ b3,
                        const __bf16* __restrict__ wsb,
                        float* __restrict__ out) {
    __shared__ __align__(16) __bf16 sAbuf[ROWS * SA_STRIDE];                  //  66,560 B
    __shared__ __align__(16) unsigned char smemB[2 * HID * SB_STRIDE * 2];    // 147,456 B
    __shared__ float sRedS[ROWS * 4];
    __shared__ float sRedQ[ROWS * 4];

    __bf16* sA  = sAbuf;
    __bf16* sB0 = (__bf16*)smemB;                         // buffer 0
    __bf16* sB1 = (__bf16*)(smemB + HID * SB_STRIDE * 2); // buffer 1
    float*  sL  = (float*)smemB;                          // logits [64][68] (in sB0)
    float*  sP  = (float*)sAbuf;                          // probs  [64][72]
    float*  sAv = (float*)sAbuf + ROWS * SP_STRIDE;       // a_bits [64][72]

    const int tid  = threadIdx.x;
    const int lane = tid & 31;
    const int wave = tid >> 5;
    const int wm   = wave & 3;
    const int wn   = wave >> 2;
    const int row0 = blockIdx.x * ROWS;

    const __bf16* w1t = wsb;
    const __bf16* w2t = wsb + BITS * HID;
    const __bf16* w3t = wsb + BITS * HID + HID * HID;

    v8f vzero = {0.f, 0.f, 0.f, 0.f, 0.f, 0.f, 0.f, 0.f};
    v8f acc[8];

    // ---- prologue: DMA W1T -> sB0, W2 chunk0 -> sB1; stage shift_bits -> sA --
    issue_w1(sB0, w1t, tid, wave);
    issue_w2_chunk(sB1, w2t, 0, tid, wave);
    for (int v = tid; v < ROWS * BITS / 4; v += NTH) {
        int r = v >> 4, c4 = (v & 15) * 4;
        float4 d = *(const float4*)(shift_bits + (size_t)(row0 + r) * BITS + c4);
        __bf16* p = sA + r * SA1_STRIDE + c4;
        p[0] = (__bf16)d.x; p[1] = (__bf16)d.y; p[2] = (__bf16)d.z; p[3] = (__bf16)d.w;
    }
    WAIT_TENSOR(1);           // W1T resident (chunk0 may still be in flight)
    __syncthreads();

    // ================= GEMM1: [64x64] x W1[64x512] =================
#pragma unroll
    for (int nt = 0; nt < 8; ++nt) acc[nt] = vzero;
#pragma unroll
    for (int ks = 0; ks < 64; ks += 32) {
        v16bf a = load_frag_a(sA, SA1_STRIDE, wm * 16, ks, lane);
#pragma unroll
        for (int nt = 0; nt < 8; ++nt) {
            v16bf b = load_frag_b(sB0, SB_STRIDE, wn * 128 + nt * 16, ks, lane);
            acc[nt] = wmma_bf16(a, b, acc[nt]);
        }
    }
    __syncthreads();                       // sB0 free
    issue_w2_chunk(sB0, w2t, 1, tid, wave);
    ln_gelu_to_sA(acc, b1, g1, be1, sA, SA_STRIDE, wm, wn, lane, sRedS, sRedQ);
    __syncthreads();

    // ====== GEMM2: [64x512] x W2[512x512], TDM-double-buffered K-chunks ======
#pragma unroll
    for (int nt = 0; nt < 8; ++nt) acc[nt] = vzero;
    for (int kc = 0; kc < 8; ++kc) {
        __bf16* bufc = (kc & 1) ? sB0 : sB1;   // chunk kc lives in buf[(kc+1)&1]
        WAIT_TENSOR(1);                         // chunk kc done; kc+1 may fly
        __syncthreads();
#pragma unroll
        for (int ks = 0; ks < 2; ++ks) {
            v16bf a = load_frag_a(sA, SA_STRIDE, wm * 16, kc * 64 + ks * 32, lane);
#pragma unroll
            for (int nt = 0; nt < 8; ++nt) {
                v16bf b = load_frag_b(bufc, SB_STRIDE, wn * 128 + nt * 16, ks * 32, lane);
                acc[nt] = wmma_bf16(a, b, acc[nt]);
            }
        }
        __syncthreads();                        // bufc free
        if (kc + 2 < 8) issue_w2_chunk(bufc, w2t, kc + 2, tid, wave);
        if (kc == 6)    issue_w3(sB1, w3t, tid, wave);  // overlap W3T with chunk7
    }
    ln_gelu_to_sA(acc, b2, g2, be2, sA, SA_STRIDE, wm, wn, lane, sRedS, sRedQ);
    __syncthreads();
    WAIT_TENSOR(0);                             // W3T resident
    __syncthreads();

    // ================= GEMM3: [64x512] x W3[512x64] =================
    v8f acc3 = vzero;
#pragma unroll 4
    for (int ks = 0; ks < 16; ++ks) {
        v16bf a = load_frag_a(sA, SA_STRIDE, wm * 16, ks * 32, lane);
        v16bf b = load_frag_b(sB1, SA_STRIDE, wn * 16, ks * 32, lane);
        acc3 = wmma_bf16(a, b, acc3);
    }
    __syncthreads();
    store_acc_lds(sL, SL_STRIDE, wm * 16, wn * 16, acc3, lane);   // -> sB0
    __syncthreads();

    // ============ softmax(logits + b3) and causal shift-convolution ==========
    {
        const int r = tid >> 3, j = tid & 7;
        float lg[8];
#pragma unroll
        for (int u = 0; u < 8; ++u) lg[u] = sL[r * SL_STRIDE + j * 8 + u] + b3[j * 8 + u];
        float mx = lg[0];
#pragma unroll
        for (int u = 1; u < 8; ++u) mx = fmaxf(mx, lg[u]);
#pragma unroll
        for (int m = 1; m < 8; m <<= 1) mx = fmaxf(mx, __shfl_xor(mx, m, 8));
        float sum = 0.f;
#pragma unroll
        for (int u = 0; u < 8; ++u) { lg[u] = expf(lg[u] - mx); sum += lg[u]; }
#pragma unroll
        for (int m = 1; m < 8; m <<= 1) sum += __shfl_xor(sum, m, 8);
        float invs = 1.0f / sum;
        const float* arow = a_bits + (size_t)(row0 + r) * BITS;
#pragma unroll
        for (int u = 0; u < 8; ++u) {
            sP[r * SP_STRIDE + j * 8 + u]  = lg[u] * invs;
            sAv[r * SP_STRIDE + j * 8 + u] = arow[j * 8 + u];
        }
        __syncthreads();
        float* orow = out + (size_t)(row0 + r) * BITS;
#pragma unroll
        for (int u = 0; u < 8; ++u) {
            int i = j * 8 + u;
            float o = 0.f;
            for (int s = 0; s <= i; ++s)
                o = fmaf(sP[r * SP_STRIDE + s], sAv[r * SP_STRIDE + (i - s)], o);
            orow[i] = o;
        }
    }
}

// ---------------------------------------------------------------------------
extern "C" void kernel_launch(void* const* d_in, const int* in_sizes, int n_in,
                              void* d_out, int out_size, void* d_ws, size_t ws_size,
                              hipStream_t stream) {
    (void)in_sizes; (void)n_in; (void)out_size; (void)ws_size;
    const float* a_bits     = (const float*)d_in[0];
    const float* shift_bits = (const float*)d_in[1];
    const float* W1  = (const float*)d_in[2];
    const float* b1  = (const float*)d_in[3];
    const float* g1  = (const float*)d_in[4];
    const float* be1 = (const float*)d_in[5];
    const float* W2  = (const float*)d_in[6];
    const float* b2  = (const float*)d_in[7];
    const float* g2  = (const float*)d_in[8];
    const float* be2 = (const float*)d_in[9];
    const float* W3  = (const float*)d_in[10];
    const float* b3  = (const float*)d_in[11];
    __bf16* wsb = (__bf16*)d_ws;    // needs 655,360 B of workspace
    float*  out = (float*)d_out;

    convert_weights_kernel<<<256, 256, 0, stream>>>(W1, W2, W3, wsb);
    fastshiftnet_fused<<<BATCH / ROWS, NTH, 0, stream>>>(
        a_bits, shift_bits, b1, g1, be1, b2, g2, be2, b3, wsb, out);
}
